// SelfAttention_72112500900459
// MI455X (gfx1250) — matrix operand order repaired and checked by reading, not compiled, not run on hardware
//
#include <hip/hip_runtime.h>
#include <hip/hip_bf16.h>

// ---------------------------------------------------------------------------
// Causal multi-head self-attention forward for MI455X (gfx1250).
// S=4096, D=1024, H=16, dk=dv=64.  bf16 WMMA (v_wmma_f32_16x16x32_bf16),
// f32 accumulate, flash-attention style online softmax.
// GEMMs stage B tiles through LDS with global_load_async_to_lds_b128
// (ASYNCcnt path) and double buffering.
// ---------------------------------------------------------------------------

#define S_TOK   4096
#define DMODEL  1024
#define HEADS   16
#define DHEAD   64

typedef __attribute__((ext_vector_type(16))) __bf16 bf16x16;
typedef __attribute__((ext_vector_type(8)))  float  f32x8;

union FragBF {
    bf16x16      v;
    unsigned int u[8];
};

union BU { __bf16 b; unsigned short s; };

#define NEG_INF (-__builtin_inff())

// A-matrix (16x32, 16-bit) K-offset for VGPR v, lane-half h (contiguous order,
// matching the B fills):  K = 16*h + 2v, 16*h + 2v + 1
__device__ __forceinline__ int a_koff(int v, int half) {
    return 16 * half + 2 * v;
}

__device__ __forceinline__ f32x8 wmma_bf16(const FragBF& a, const FragBF& b, f32x8 c) {
    return __builtin_amdgcn_wmma_f32_16x16x32_bf16(
        /*neg_a=*/false, a.v, /*neg_b=*/false, b.v,
        /*c_mod=*/(short)0, c, /*reuse_a=*/false, /*reuse_b=*/false);
}

// async copy 16B global -> LDS (per active lane), tracked by ASYNCcnt
__device__ __forceinline__ void async_ld_b128(unsigned lds_byte, unsigned voff_byte,
                                              const __bf16* base) {
    asm volatile("global_load_async_to_lds_b128 %0, %1, %2"
                 :: "v"(lds_byte), "v"(voff_byte),
                    "s"((unsigned long long)(uintptr_t)base)
                 : "memory");
}

// reductions within a 16-lane half of the wave (rows of the C-layout tile)
__device__ __forceinline__ float half_max16(float v) {
    #pragma unroll
    for (int m = 1; m < 16; m <<= 1) v = fmaxf(v, __shfl_xor(v, m, 32));
    return v;
}
__device__ __forceinline__ float half_sum16(float v) {
    #pragma unroll
    for (int m = 1; m < 16; m <<= 1) v += __shfl_xor(v, m, 32);
    return v;
}

// ---------------------------------------------------------------------------
// f32 -> bf16 cast, 4 elements / thread (b128 load, b64 store)
// ---------------------------------------------------------------------------
__global__ void __launch_bounds__(256)
cast_f32_bf16_x4(const float* __restrict__ in, __bf16* __restrict__ out, int n4) {
    int i = blockIdx.x * blockDim.x + threadIdx.x;
    if (i >= n4) return;
    float4 f = ((const float4*)in)[i];
    BU b0, b1, b2, b3;
    b0.b = (__bf16)f.x; b1.b = (__bf16)f.y; b2.b = (__bf16)f.z; b3.b = (__bf16)f.w;
    ((ushort4*)out)[i] = make_ushort4(b0.s, b1.s, b2.s, b3.s);
}

// ---------------------------------------------------------------------------
// Generic NT GEMM:  C[M,N] = A[M,K] * B[N,K]^T    (A, B bf16 row-major)
//   grid.x = M/128, grid.y = N/64, grid.z = head (B += z*bhs, C += z*chs)
//   256 threads = 8 waves; wave w owns rows [blk*128 + w*16, +16) x 64 cols.
//   B tile (64 rows x 32 K = 4 KB) staged to LDS once per workgroup via
//   global_load_async_to_lds_b128 (exactly 1 async b128 per thread per step),
//   double-buffered so the copy overlaps the WMMAs.
//   OUT_BF16: store bf16 (else f32).  TRANS: store C^T (for V -> Vt).
// ---------------------------------------------------------------------------
template <int OUT_BF16, int TRANS>
__global__ void __launch_bounds__(256)
gemm_nt(const __bf16* __restrict__ A, int lda,
        const __bf16* __restrict__ B, int ldb, size_t bhs,
        void* __restrict__ C, int ldc, size_t chs, int Kdim) {
    __shared__ __bf16 sB[2][64][32];       // 2 x 4 KB double buffer

    const int tid   = threadIdx.x;
    const int lane  = tid & 31;
    const int w     = tid >> 5;
    const int half  = lane >> 4;
    const int n16   = lane & 15;
    const int mbase = blockIdx.x * 128 + w * 16;
    const int nbase = blockIdx.y * 64;
    const __bf16* Bh = B + (size_t)blockIdx.z * bhs + (size_t)nbase * ldb;

    // this thread's async chunk of the 64x32 B tile: row = tid/4, 8 elems
    const int brow = tid >> 2;
    const int bk8  = (tid & 3) * 8;
    const unsigned voff      = (unsigned)(brow * ldb + bk8) * 2u;        // global bytes
    const unsigned lds_base  = (unsigned)(uintptr_t)&sB[0][0][0];
    const unsigned lds_chunk = (unsigned)(brow * 32 + bk8) * 2u;         // LDS bytes

    f32x8 acc[4];
    const f32x8 vzero = {0.f, 0.f, 0.f, 0.f, 0.f, 0.f, 0.f, 0.f};
    #pragma unroll
    for (int j = 0; j < 4; ++j) acc[j] = vzero;

    const __bf16* arow = A + (size_t)(mbase + n16) * lda;

    // prologue: start filling buffer 0 with K-chunk 0
    async_ld_b128(lds_base + lds_chunk, voff, Bh);

    int buf = 0;
    for (int k0 = 0; k0 < Kdim; k0 += 32) {
        asm volatile("s_wait_asynccnt 0x0" ::: "memory");  // my tile chunk landed
        __syncthreads();                                    // whole tile landed
        if (k0 + 32 < Kdim)                                 // refill other buffer
            async_ld_b128(lds_base + (unsigned)((buf ^ 1) * 4096) + lds_chunk,
                          voff, Bh + (k0 + 32));

        FragBF af;
        #pragma unroll
        for (int v = 0; v < 8; ++v)
            af.u[v] = *(const unsigned int*)(arow + k0 + a_koff(v, half));
        __builtin_prefetch(arow + k0 + 128, 0, 3);

        #pragma unroll
        for (int j = 0; j < 4; ++j) {
            FragBF bfr;                                     // shared tile: ds_load
            #pragma unroll
            for (int v = 0; v < 8; ++v)
                bfr.u[v] = *(const unsigned int*)&sB[buf][j * 16 + n16][16 * half + 2 * v];
            acc[j] = wmma_bf16(af, bfr, acc[j]);
        }
        buf ^= 1;
    }

    #pragma unroll
    for (int j = 0; j < 4; ++j) {
        #pragma unroll
        for (int v = 0; v < 8; ++v) {
            float val = acc[j][v];
            int m  = mbase + v + 8 * half;      // C layout: M = v + 8*half
            int nl = nbase + j * 16 + n16;      //           N = lane&15
            if (OUT_BF16) {
                __bf16* Cb = (__bf16*)C + (size_t)blockIdx.z * chs;
                if (TRANS) Cb[(size_t)nl * ldc + m] = (__bf16)val;
                else       Cb[(size_t)m * ldc + nl] = (__bf16)val;
            } else {
                ((float*)C)[(size_t)m * ldc + nl] = val;
            }
        }
    }
}

// ---------------------------------------------------------------------------
// Flash attention core: grid(S/64, H), 128 threads = 4 waves.
// Wave w: 16 query rows [qbase, qbase+16) of head h. Online softmax over
// 32-key blocks.  Q,K: bf16 [H][S][64];  Vt: bf16 [H][64][S];
// output mh: bf16 [S][H*64].  (Waves have different trip counts, so no
// workgroup barriers here -- per-wave LDS bounce only.)
// ---------------------------------------------------------------------------
__global__ void __launch_bounds__(128)
attn_kernel(const __bf16* __restrict__ Qb, const __bf16* __restrict__ Kb,
            const __bf16* __restrict__ Vtb, __bf16* __restrict__ mhb) {
    __shared__ __bf16 lds_p[4][16][32];   // 1 KB per wave: P tile bounce buffer

    const int lane  = threadIdx.x & 31;
    const int w     = threadIdx.x >> 5;
    const int half  = lane >> 4;
    const int n16   = lane & 15;
    const int h     = blockIdx.y;
    const int qbase = blockIdx.x * 64 + w * 16;

    // Q fragments for this wave's 16 rows (dk=64 -> two 16x32 A-frags)
    FragBF qf[2];
    const __bf16* qrow = Qb + (size_t)(h * S_TOK + qbase + n16) * DHEAD;
    #pragma unroll
    for (int kc = 0; kc < 2; ++kc)
        #pragma unroll
        for (int v = 0; v < 8; ++v)
            qf[kc].u[v] = *(const unsigned int*)(qrow + kc * 32 + a_koff(v, half));

    float m_i[8], l_i[8];
    f32x8 o[4];
    const f32x8 vzero = {0.f, 0.f, 0.f, 0.f, 0.f, 0.f, 0.f, 0.f};
    #pragma unroll
    for (int v = 0; v < 8; ++v) { m_i[v] = NEG_INF; l_i[v] = 0.f; }
    #pragma unroll
    for (int j = 0; j < 4; ++j) o[j] = vzero;

    const int tmax = (qbase + 15) >> 5;   // last 32-key block with unmasked work
    for (int tb = 0; tb <= tmax; ++tb) {
        // ---- S = Q K^T for a 16x32 score tile (two 16x16 col sub-tiles) ----
        f32x8 sc[2] = {vzero, vzero};
        #pragma unroll
        for (int sub = 0; sub < 2; ++sub) {
            const int t0 = tb * 32 + sub * 16;
            #pragma unroll
            for (int kc = 0; kc < 2; ++kc) {
                FragBF bk;
                const __bf16* kr =
                    Kb + (size_t)(h * S_TOK + t0 + n16) * DHEAD + kc * 32 + 16 * half;
                #pragma unroll
                for (int v = 0; v < 8; ++v)
                    bk.u[v] = *(const unsigned int*)(kr + 2 * v);
                sc[sub] = wmma_bf16(qf[kc], bk, sc[sub]);
            }
        }

        // ---- online softmax update (rows = v + 8*half, cols across lanes) --
        float p0s[8], p1s[8];
        #pragma unroll
        for (int v = 0; v < 8; ++v) {
            const int qrow_g = qbase + v + 8 * half;
            float s0 = sc[0][v] * 0.125f;               // 1/sqrt(64)
            float s1 = sc[1][v] * 0.125f;
            const int k0 = tb * 32 + n16;
            if (k0      > qrow_g) s0 = NEG_INF;          // causal mask
            if (k0 + 16 > qrow_g) s1 = NEG_INF;
            float bm = half_max16(fmaxf(s0, s1));
            float mn = fmaxf(m_i[v], bm);
            float alpha = __expf(m_i[v] - mn);
            float p0 = __expf(s0 - mn);
            float p1 = __expf(s1 - mn);
            float ps = half_sum16(p0 + p1);
            l_i[v] = l_i[v] * alpha + ps;
            m_i[v] = mn;
            p0s[v] = p0; p1s[v] = p1;
            #pragma unroll
            for (int j = 0; j < 4; ++j) o[j][v] *= alpha;
        }

        // ---- C-layout P -> A-layout bf16 fragment via per-wave LDS bounce --
        #pragma unroll
        for (int v = 0; v < 8; ++v) {
            lds_p[w][v + 8 * half][n16]      = (__bf16)p0s[v];
            lds_p[w][v + 8 * half][16 + n16] = (__bf16)p1s[v];
        }
        asm volatile("s_wait_dscnt 0x0" ::: "memory");   // LDS stores visible to this wave
        FragBF pf;
        #pragma unroll
        for (int v = 0; v < 8; ++v)
            pf.u[v] = *(const unsigned int*)&lds_p[w][n16][a_koff(v, half)];

        // ---- O += P * V (Vt pre-transposed: contiguous K pairs) ------------
        #pragma unroll
        for (int j = 0; j < 4; ++j) {
            FragBF bv;
            const __bf16* vr =
                Vtb + (size_t)(h * DHEAD + j * 16 + n16) * S_TOK + tb * 32 + 16 * half;
            #pragma unroll
            for (int v = 0; v < 8; ++v)
                bv.u[v] = *(const unsigned int*)(vr + 2 * v);
            o[j] = wmma_bf16(pf, bv, o[j]);
        }
    }

    // ---- epilogue: normalize, store bf16 into multihead [S, H*64] ----------
    #pragma unroll
    for (int j = 0; j < 4; ++j) {
        #pragma unroll
        for (int v = 0; v < 8; ++v) {
            float val = o[j][v] / l_i[v];
            int srow = qbase + v + 8 * half;
            int col  = h * DHEAD + j * 16 + n16;
            mhb[(size_t)srow * DMODEL + col] = (__bf16)val;
        }
    }
}

// ---------------------------------------------------------------------------
// Launcher
// ---------------------------------------------------------------------------
extern "C" void kernel_launch(void* const* d_in, const int* in_sizes, int n_in,
                              void* d_out, int out_size, void* d_ws, size_t ws_size,
                              hipStream_t stream) {
    const float* x  = (const float*)d_in[0];   // [S, D]
    const float* Wq = (const float*)d_in[1];   // [H, dk, D]
    const float* Wk = (const float*)d_in[2];   // [H, dk, D]
    const float* Wv = (const float*)d_in[3];   // [H, dv, D]
    const float* Wo = (const float*)d_in[4];   // [D, H*dv]

    char* ws = (char*)d_ws;
    const size_t MB = 1u << 20;
    __bf16* xb  = (__bf16*)(ws + 0 * MB);   //  8 MB  [S, D]
    __bf16* Wqb = (__bf16*)(ws + 8 * MB);   //  2 MB  [H, 64, D]
    __bf16* Wkb = (__bf16*)(ws + 10 * MB);  //  2 MB
    __bf16* Wvb = (__bf16*)(ws + 12 * MB);  //  2 MB
    __bf16* Wob = (__bf16*)(ws + 14 * MB);  //  2 MB  [D, H*64]
    __bf16* Qb  = (__bf16*)(ws + 16 * MB);  //  8 MB  [H, S, 64]
    __bf16* Kb  = (__bf16*)(ws + 24 * MB);  //  8 MB  [H, S, 64]
    __bf16* Vtb = (__bf16*)(ws + 32 * MB);  //  8 MB  [H, 64, S]
    __bf16* mhb = (__bf16*)(ws + 40 * MB);  //  8 MB  [S, H*64]

    const int nX = S_TOK * DMODEL;           // 4M
    const int nW = HEADS * DHEAD * DMODEL;   // 1M

    cast_f32_bf16_x4<<<dim3((nX / 4 + 255) / 256), 256, 0, stream>>>(x,  xb,  nX / 4);
    cast_f32_bf16_x4<<<dim3((nW / 4 + 255) / 256), 256, 0, stream>>>(Wq, Wqb, nW / 4);
    cast_f32_bf16_x4<<<dim3((nW / 4 + 255) / 256), 256, 0, stream>>>(Wk, Wkb, nW / 4);
    cast_f32_bf16_x4<<<dim3((nW / 4 + 255) / 256), 256, 0, stream>>>(Wv, Wvb, nW / 4);
    cast_f32_bf16_x4<<<dim3((nW / 4 + 255) / 256), 256, 0, stream>>>(Wo, Wob, nW / 4);

    // Q[h] = x @ Wq[h]^T   (M=S, N=64 per head, K=D)
    gemm_nt<1, 0><<<dim3(S_TOK / 128, 1, HEADS), 256, 0, stream>>>(
        xb, DMODEL, Wqb, DMODEL, (size_t)DHEAD * DMODEL,
        (void*)Qb, DHEAD, (size_t)S_TOK * DHEAD, DMODEL);
    gemm_nt<1, 0><<<dim3(S_TOK / 128, 1, HEADS), 256, 0, stream>>>(
        xb, DMODEL, Wkb, DMODEL, (size_t)DHEAD * DMODEL,
        (void*)Kb, DHEAD, (size_t)S_TOK * DHEAD, DMODEL);
    // V stored transposed: Vt[h][vcol][s]
    gemm_nt<1, 1><<<dim3(S_TOK / 128, 1, HEADS), 256, 0, stream>>>(
        xb, DMODEL, Wvb, DMODEL, (size_t)DHEAD * DMODEL,
        (void*)Vtb, S_TOK, (size_t)DHEAD * S_TOK, DMODEL);

    attn_kernel<<<dim3(S_TOK / 64, HEADS), 128, 0, stream>>>(Qb, Kb, Vtb, mhb);

    // out = mh @ Wo^T   (M=S, N=D, K=H*dv) -> f32 d_out
    gemm_nt<0, 0><<<dim3(S_TOK / 128, DMODEL / 64, 1), 256, 0, stream>>>(
        mhb, DMODEL, Wob, DMODEL, (size_t)0,
        d_out, DMODEL, (size_t)0, DMODEL);
}
